// AGNN_40853728920168
// MI455X (gfx1250) — matrix operand order repaired and checked by reading, not compiled, not run on hardware
//
#include <hip/hip_runtime.h>
#include <math.h>
#include <stdint.h>

// ---------------- problem constants ----------------
static const int B_ = 4, T_ = 40, N_ = 2912, F_ = 161, E_ = N_ * 6, BT_ = 160;
static const long long BTN_ = (long long)BT_ * N_;   // 465920

typedef float v8f __attribute__((ext_vector_type(8)));
typedef __bf16 v16bf __attribute__((ext_vector_type(16)));

#define GS(i, n) for (long long i = (long long)blockIdx.x * blockDim.x + threadIdx.x; \
                      i < (long long)(n); i += (long long)gridDim.x * blockDim.x)

static __device__ __forceinline__ unsigned short f2bf(float f) {
  union { float f; unsigned u; } a; a.f = f;
  unsigned u = a.u;
  unsigned r = u + 0x7FFFu + ((u >> 16) & 1u);   // round-to-nearest-even
  return (unsigned short)(r >> 16);
}
static __device__ __forceinline__ float sigf(float x) { return 1.f / (1.f + expf(-x)); }

// ---------------- generic utilities ----------------
__global__ void fill_kernel(float* p, float v, long long n) { GS(i, n) p[i] = v; }
__global__ void copy_kernel(float* dst, const float* src, long long n) { GS(i, n) dst[i] = src[i]; }

// ---------------- bn1: per-node stats fused as scale/shift ----------------
__global__ void bn1_stats_kernel(const float* __restrict__ x, const float* __restrict__ g,
                                 const float* __restrict__ b, float* __restrict__ ascale,
                                 float* __restrict__ ashift) {
  int n = blockIdx.x, tid = threadIdx.x;
  __shared__ float s1[256], s2[256];
  const long long cnt = (long long)BT_ * F_;
  float a = 0.f, q = 0.f;
  for (long long i = tid; i < cnt; i += 256) {
    long long bt = i / F_; int f = (int)(i % F_);
    float v = x[(bt * N_ + n) * (long long)F_ + f];
    a += v; q += v * v;
  }
  s1[tid] = a; s2[tid] = q; __syncthreads();
  for (int s = 128; s > 0; s >>= 1) {
    if (tid < s) { s1[tid] += s1[tid + s]; s2[tid] += s2[tid + s]; }
    __syncthreads();
  }
  if (tid == 0) {
    float m = s1[0] / (float)cnt;
    float var = s2[0] / (float)cnt - m * m;
    float inv = rsqrtf(var + 1e-5f) * g[n];
    ascale[n] = inv; ashift[n] = b[n] - m * inv;
  }
}

// ---------------- GCN adjacency -> CSR ----------------
__global__ void deg_count_kernel(const int* col, float* deg) { GS(i, E_) atomicAdd(&deg[col[i]], 1.f); }
__global__ void dinv_kernel(const float* deg, float* dinv) { GS(i, N_) dinv[i] = rsqrtf(deg[i] + 1.f); }
__global__ void edge_norm_kernel(const int* row, const int* col, const float* dinv, float* nrm) {
  GS(i, E_) nrm[i] = dinv[row[i]] * dinv[col[i]];
}
__global__ void scan_kernel(const float* deg, int* offs) {
  if (blockIdx.x == 0 && threadIdx.x == 0) {
    int a = 0;
    for (int m = 0; m < N_; ++m) { offs[m] = a; a += (int)(deg[m] + 0.5f); }
    offs[N_] = a;
  }
}
__global__ void csr_fill_kernel(const int* row, const int* col, const float* nrm,
                                const int* offs, int* cur, int* csrR, float* csrW) {
  GS(i, E_) {
    int m = col[i];
    int p = atomicAdd(&cur[m], 1);
    int pos = offs[m] + p;
    csrR[pos] = row[i]; csrW[pos] = nrm[i];
  }
}
// out[bt,m,f] = dinv[m]^2*in[bt,m,f] + sum_j w_j * in[bt,row_j,f]
__global__ void spmm_kernel(const float* __restrict__ in, float* __restrict__ out,
                            const int* __restrict__ offs, const int* __restrict__ csrR,
                            const float* __restrict__ csrW, const float* __restrict__ dinv) {
  int m = blockIdx.x, bt = blockIdx.y, f = threadIdx.x;
  if (f >= F_) return;
  long long base = (long long)bt * N_ * F_;
  float s = dinv[m] * dinv[m] * in[base + (long long)m * F_ + f];
  int st = offs[m], en = offs[m + 1];
  for (int j = st; j < en; ++j) s += csrW[j] * in[base + (long long)csrR[j] * F_ + f];
  out[base + (long long)m * F_ + f] = s;
}

// ---------------- aW = W - W^T - 0.1 I ----------------
__global__ void make_aw_kernel(const float* __restrict__ W, float* __restrict__ aw) {
  GS(i, (long long)F_ * F_) {
    int r = (int)(i / F_), c = (int)(i % F_);
    aw[i] = W[i] - W[(long long)c * F_ + r] - (r == c ? 0.1f : 0.f);
  }
}

// ---------------- pack weight into wave32 WMMA B-fragment layout ----------------
// fragment element (kt,nt,lane,j): n = nt*16 + (lane&15); k = kt*32 + (lane>=16 ? 16+j : j)
// Ntp is padded to an even tile count; n >= Nreal packs zero.
__global__ void pack_b_kernel(const float* __restrict__ W, int wld, int Kreal, int Nreal,
                              int trans, int Ktiles, int Ntp, unsigned short* __restrict__ out) {
  long long total = (long long)Ktiles * Ntp * 512;
  GS(i, total) {
    int j = (int)(i & 15);
    int lane = (int)((i >> 4) & 31);
    long long tile = i >> 9;
    int nt = (int)(tile % Ntp), kt = (int)(tile / Ntp);
    int n = nt * 16 + (lane & 15);
    int k = kt * 32 + ((lane >> 4) ? 16 + j : j);
    float v = 0.f;
    if (k < Kreal && n < Nreal)
      v = trans ? W[(long long)n * wld + k] : W[(long long)k * wld + n];
    out[i] = f2bf(v);
  }
}

// ---------------- WMMA bf16 GEMM: C = act(A*B [+bias]) ----------------
// Block = 4 waves x 16 rows = 64 rows, and TWO 16-col N-tiles per block.
// Per k-tile the 2KB packed-B fragment pair is staged to LDS with one
// global_load_async_to_lds_b128 per thread and shared by all 4 waves; each
// wave then issues two v_wmma_f32_16x16x32_bf16 reusing its A fragment.
// mode 0: acc+bias ; mode 1: tanh(acc+bias)
// mode 2 (anti-conv epilogue): tanh(xin + eps*tanh(acc + Hsum + bias))
__global__ void __launch_bounds__(128)
wmma_gemm_kernel(const float* __restrict__ A, long long lda, long long M,
                 int Kreal, int Ktiles,
                 const unsigned short* __restrict__ packB, int Ntp, int Ncols,
                 const float* __restrict__ bias,
                 float* __restrict__ C, long long ldc,
                 int mode, float epsv,
                 const float* __restrict__ ascale, const float* __restrict__ ashift, int nmod,
                 const float* __restrict__ Xin, const float* __restrict__ Hsum) {
  __shared__ __align__(32) unsigned short sB[1024];
  const int tid = threadIdx.x;
  const int lane = tid & 31;
  const int wv = tid >> 5;
  const int half = lane >> 4;
  const int nt0 = blockIdx.y * 2;
  long long m0 = ((long long)blockIdx.x * 4 + wv) << 4;
  long long arow = m0 + (lane & 15);
  if (arow >= M) arow = M - 1;
  const float* Arow = A + arow * lda;
  float asc = 1.f, ash = 0.f;
  if (ascale) { int nn = (int)(arow % nmod); asc = ascale[nn]; ash = ashift[nn]; }

  union FragA { v16bf v; unsigned short s[16]; } fa;
  v8f acc0, acc1;
#pragma unroll
  for (int i = 0; i < 8; ++i) { acc0[i] = 0.f; acc1[i] = 0.f; }

  for (int kt = 0; kt < Ktiles; ++kt) {
    // ---- async stage of the two shared B fragments: 128 thr x 16B = 2KB ----
    {
      const unsigned short* gb = packB + ((long long)kt * Ntp + nt0) * 512;
      unsigned ldsAddr = (unsigned)(uintptr_t)(&sB[0]) + (unsigned)tid * 16u;
      unsigned long long gaddr = (unsigned long long)(uintptr_t)((const char*)gb + tid * 16);
      asm volatile("global_load_async_to_lds_b128 %0, %1, off"
                   :: "v"(ldsAddr), "v"(gaddr) : "memory");
      asm volatile("s_wait_asynccnt 0" ::: "memory");
    }
    __syncthreads();

    const int kb = kt * 32;
    const int k1 = kb + half * 16;       // A layout: lanes<16 -> K [kb..kb+7],[kb+8..kb+15]
    const int k2 = kb + 8 + half * 16;   //           lanes>=16 -> K [kb+16..23],[kb+24..31]
    if (kb + 32 <= Kreal) {
      // fast path: unguarded contiguous 8-element runs (coalesced b32 bursts)
#pragma unroll
      for (int j = 0; j < 8; ++j) fa.s[j] = f2bf(Arow[k1 + j] * asc + ash);
#pragma unroll
      for (int j = 0; j < 8; ++j) fa.s[8 + j] = f2bf(Arow[k2 + j] * asc + ash);
    } else {
      // tail k-tile: clamped-address loads + select (no exec-masked loads)
#pragma unroll
      for (int j = 0; j < 8; ++j) {
        int k = k1 + j;
        int kc = k < Kreal ? k : 0;
        float v = Arow[kc] * asc + ash;
        fa.s[j] = f2bf(k < Kreal ? v : 0.f);
      }
#pragma unroll
      for (int j = 0; j < 8; ++j) {
        int k = k2 + j;
        int kc = k < Kreal ? k : 0;
        float v = Arow[kc] * asc + ash;
        fa.s[8 + j] = f2bf(k < Kreal ? v : 0.f);
      }
    }
    v16bf fb0 = ((const v16bf*)sB)[lane];
    v16bf fb1 = ((const v16bf*)sB)[32 + lane];
    acc0 = __builtin_amdgcn_wmma_f32_16x16x32_bf16(false, fa.v, false, fb0,
                                                   (short)0, acc0, false, false);
    acc1 = __builtin_amdgcn_wmma_f32_16x16x32_bf16(false, fa.v, false, fb1,
                                                   (short)0, acc1, false, false);
    __syncthreads();
  }

  v8f accs[2] = {acc0, acc1};
#pragma unroll
  for (int tt = 0; tt < 2; ++tt) {
    const int col = (nt0 + tt) * 16 + (lane & 15);
    const float bcol = (bias && col < Ncols) ? bias[col] : 0.f;
#pragma unroll
    for (int g = 0; g < 8; ++g) {
      long long row = m0 + half * 8 + g;
      if (row < M && col < Ncols) {
        long long off = row * ldc + col;
        float v;
        if (mode == 2) {
          float xin = Xin[off];
          if (ascale) { int nn = (int)(row % nmod); xin = xin * ascale[nn] + ashift[nn]; }
          v = tanhf(xin + epsv * tanhf(accs[tt][g] + Hsum[off] + bcol));
        } else if (mode == 1) {
          v = tanhf(accs[tt][g] + bcol);
        } else {
          v = accs[tt][g] + bcol;
        }
        C[off] = v;
      }
    }
  }
}

// ---------------- l40: reduce over T with tanh ----------------
__global__ void l40_kernel(const float* __restrict__ xf, const float* __restrict__ w40,
                           const float* __restrict__ b40, float* __restrict__ xs) {
  GS(i, (long long)B_ * N_ * 50) {
    int g = (int)(i % 50);
    long long bn = i / 50;
    int n = (int)(bn % N_), b = (int)(bn / N_);
    float s = b40[0];
    for (int t = 0; t < T_; ++t)
      s += xf[(((long long)b * T_ + t) * N_ + n) * 50 + g] * w40[t];
    xs[i] = tanhf(s);
  }
}

// ---------------- LSTM (scan over B steps) ----------------
__global__ void lstm_gates_kernel(const float* __restrict__ xs, const float* __restrict__ h,
                                  const float* __restrict__ Wih, const float* __restrict__ Whh,
                                  const float* __restrict__ bih, const float* __restrict__ bhh,
                                  float* __restrict__ gates) {
  GS(i, (long long)N_ * 120) {
    int u = (int)(i % 120), n = (int)(i / 120);
    float s = bih[u] + bhh[u];
    const float* xr = xs + (long long)n * 50;
    const float* wr = Wih + (long long)u * 50;
    for (int j = 0; j < 50; ++j) s += xr[j] * wr[j];
    const float* hr = h + (long long)n * 30;
    const float* vr = Whh + (long long)u * 30;
    for (int j = 0; j < 30; ++j) s += hr[j] * vr[j];
    gates[i] = s;
  }
}
__global__ void lstm_update_kernel(const float* __restrict__ gates, float* __restrict__ h,
                                   float* __restrict__ c) {
  GS(i, (long long)N_ * 30) {
    int u = (int)(i % 30), n = (int)(i / 30);
    const float* g = gates + (long long)n * 120;
    float gi = sigf(g[u]), gf = sigf(g[30 + u]), gg = tanhf(g[60 + u]), go = sigf(g[90 + u]);
    float cc = gf * c[i] + gi * gg;
    c[i] = cc;
    h[i] = go * tanhf(cc);
  }
}

// ---------------- small dense FC ----------------
__global__ void fc_kernel(const float* __restrict__ in, int Kin, const float* __restrict__ W,
                          const float* __restrict__ bias, float* __restrict__ out,
                          int Nout, long long R, int act) {
  GS(i, R * (long long)Nout) {
    int o = (int)(i % Nout);
    long long r = i / Nout;
    float s = bias ? bias[o] : 0.f;
    const float* ir = in + r * (long long)Kin;
    const float* wr = W + (long long)o * Kin;
    for (int j = 0; j < Kin; ++j) s += ir[j] * wr[j];
    out[i] = act ? tanhf(s) : s;
  }
}

// ---------------- global batch-norm (scalar stats) ----------------
__global__ void gstats_kernel(const float* __restrict__ x, long long n, float* acc) {
  __shared__ float s1[256], s2[256];
  float a = 0.f, q = 0.f;
  GS(i, n) { float v = x[i]; a += v; q += v * v; }
  s1[threadIdx.x] = a; s2[threadIdx.x] = q; __syncthreads();
  for (int s = 128; s > 0; s >>= 1) {
    if (threadIdx.x < s) { s1[threadIdx.x] += s1[threadIdx.x + s]; s2[threadIdx.x] += s2[threadIdx.x + s]; }
    __syncthreads();
  }
  if (!threadIdx.x) { atomicAdd(&acc[0], s1[0]); atomicAdd(&acc[1], s2[0]); }
}
__global__ void gbn_kernel(float* __restrict__ x, long long n, const float* __restrict__ acc,
                           const float* __restrict__ g, const float* __restrict__ b, int act) {
  float m = acc[0] / (float)n;
  float var = acc[1] / (float)n - m * m;
  float inv = rsqrtf(var + 1e-5f) * g[0];
  float sh = b[0] - m * inv;
  GS(i, n) { float v = x[i] * inv + sh; x[i] = act ? tanhf(v) : v; }
}

// ---------------- kNN top-6 -> binary adjacency ----------------
__global__ void sqn_kernel(const float* __restrict__ emb, float* __restrict__ sq) {
  GS(i, N_) {
    const float* e = emb + i * 30;
    float s = 0.f;
    for (int j = 0; j < 30; ++j) s += e[j] * e[j];
    sq[i] = s;
  }
}
__global__ void topk_kernel(const float* __restrict__ emb, const float* __restrict__ sq,
                            int* __restrict__ idxOut, float* __restrict__ Wadj) {
  const int n = blockIdx.x, tid = threadIdx.x;   // 128 threads
  __shared__ float se[30];
  __shared__ float cd[128 * 6];
  __shared__ int ci[128 * 6];
  if (tid < 30) se[tid] = emb[(long long)n * 30 + tid];
  __syncthreads();
  float bd[6]; int bi[6];
#pragma unroll
  for (int j = 0; j < 6; ++j) { bd[j] = 3.0e38f; bi[j] = 0; }
  const float sn = sq[n];
  for (int m = tid; m < N_; m += 128) {
    if (m == n) continue;
    const float* em = emb + (long long)m * 30;
    float dot = 0.f;
#pragma unroll
    for (int j = 0; j < 30; ++j) dot += se[j] * em[j];
    float d2 = sn + sq[m] - 2.f * dot;
    if (d2 < bd[5]) {
      int p = 5;
      while (p > 0 && bd[p - 1] > d2) { bd[p] = bd[p - 1]; bi[p] = bi[p - 1]; --p; }
      bd[p] = d2; bi[p] = m;
    }
  }
#pragma unroll
  for (int j = 0; j < 6; ++j) { cd[tid * 6 + j] = bd[j]; ci[tid * 6 + j] = bi[j]; }
  __syncthreads();
  if (tid == 0) {
    for (int s = 0; s < 6; ++s) {
      float best = 3.1e38f; int bq = 0;
      for (int q = 0; q < 128 * 6; ++q)
        if (cd[q] < best) { best = cd[q]; bq = q; }
      int m = ci[bq]; cd[bq] = 3.2e38f;
      idxOut[n * 6 + s] = m;
      Wadj[(long long)n * N_ + m] = 1.0f;
    }
  }
}

// ---------------- w = softmax(-50 e^{-8w}) - softmax(-50 e^{8w}) ----------------
__global__ void wv_kernel(const float* __restrict__ mu, const int* __restrict__ idx,
                          float* __restrict__ wv) {
  GS(i, (long long)B_ * N_) {
    int n = (int)(i % N_), b = (int)(i / N_);
    float s = 0.f;
    for (int j = 0; j < 6; ++j) s += mu[(long long)b * N_ + idx[n * 6 + j]];
    wv[i] = s;
  }
}
__global__ void wsoftmax_kernel(const float* __restrict__ wv, float* __restrict__ w) {
  int b = blockIdx.x, tid = threadIdx.x;   // 256 threads
  __shared__ float red[256];
  __shared__ float sm1, sm2, ss1, ss2;
  float m1 = -3.4e38f, m2 = -3.4e38f;
  for (int n = tid; n < N_; n += 256) {
    float x = wv[b * N_ + n];
    m1 = fmaxf(m1, -50.f * expf(-8.f * x));
    m2 = fmaxf(m2, -50.f * expf(8.f * x));
  }
  red[tid] = m1; __syncthreads();
  for (int s = 128; s > 0; s >>= 1) { if (tid < s) red[tid] = fmaxf(red[tid], red[tid + s]); __syncthreads(); }
  if (!tid) sm1 = red[0];
  __syncthreads();
  red[tid] = m2; __syncthreads();
  for (int s = 128; s > 0; s >>= 1) { if (tid < s) red[tid] = fmaxf(red[tid], red[tid + s]); __syncthreads(); }
  if (!tid) sm2 = red[0];
  __syncthreads();
  float a1 = 0.f, a2 = 0.f;
  for (int n = tid; n < N_; n += 256) {
    float x = wv[b * N_ + n];
    a1 += expf(-50.f * expf(-8.f * x) - sm1);
    a2 += expf(-50.f * expf(8.f * x) - sm2);
  }
  red[tid] = a1; __syncthreads();
  for (int s = 128; s > 0; s >>= 1) { if (tid < s) red[tid] += red[tid + s]; __syncthreads(); }
  if (!tid) ss1 = red[0];
  __syncthreads();
  red[tid] = a2; __syncthreads();
  for (int s = 128; s > 0; s >>= 1) { if (tid < s) red[tid] += red[tid + s]; __syncthreads(); }
  if (!tid) ss2 = red[0];
  __syncthreads();
  for (int n = tid; n < N_; n += 256) {
    float x = wv[b * N_ + n];
    w[b * N_ + n] = expf(-50.f * expf(-8.f * x) - sm1) / ss1 -
                    expf(-50.f * expf(8.f * x) - sm2) / ss2;
  }
}

// ---------------- R[b,t] = sum_n r[b,t,n]*w[b,n] ----------------
__global__ void r_reduce_kernel(const float* __restrict__ r, const float* __restrict__ w,
                                float* __restrict__ R) {
  int bt = blockIdx.x, b = bt / T_;
  __shared__ float red[256];
  float s = 0.f;
  for (int n = threadIdx.x; n < N_; n += 256)
    s += r[(long long)bt * N_ + n] * w[(long long)b * N_ + n];
  red[threadIdx.x] = s; __syncthreads();
  for (int st = 128; st > 0; st >>= 1) { if (threadIdx.x < st) red[threadIdx.x] += red[threadIdx.x + st]; __syncthreads(); }
  if (!threadIdx.x) R[bt] = red[0];
}

// ---------------- pre_r[b,t,n] = beta[b,n]*R[b,t] ----------------
__global__ void prer_kernel(const float* __restrict__ beta, const float* __restrict__ R,
                            float* __restrict__ out) {
  GS(i, BTN_) {
    int n = (int)(i % N_);
    long long bt = i / N_;
    int b = (int)(bt / T_);
    out[i] = beta[(long long)b * N_ + n] * R[bt];
  }
}

// ---------------- host side ----------------
static inline dim3 gsg(long long n) {
  long long b = (n + 255) / 256;
  if (b > 262144) b = 262144;
  return dim3((unsigned)b);
}
static inline void launch_gemm(hipStream_t s, const float* A, long long lda, long long M,
                               int Kreal, int Ktiles, const unsigned short* pB, int Ntp,
                               int Ncols, const float* bias, float* C, long long ldc,
                               int mode, float epsv, const float* asc, const float* ash,
                               int nmod, const float* Xin, const float* Hsum) {
  long long mt = (M + 15) / 16;
  dim3 grid((unsigned)((mt + 3) / 4), (unsigned)(Ntp / 2));
  wmma_gemm_kernel<<<grid, 128, 0, s>>>(A, lda, M, Kreal, Ktiles, pB, Ntp, Ncols, bias,
                                        C, ldc, mode, epsv, asc, ash, nmod, Xin, Hsum);
}
static inline float* alignup(float* p) {
  return (float*)(((uintptr_t)p + 63) & ~(uintptr_t)63);
}

extern "C" void kernel_launch(void* const* d_in, const int* in_sizes, int n_in,
                              void* d_out, int out_size, void* d_ws, size_t ws_size,
                              hipStream_t stream) {
  (void)in_sizes; (void)n_in; (void)out_size; (void)ws_size;
  const float* d_x = (const float*)d_in[0];
  const float* d_r = (const float*)d_in[1];
  const float* d_h = (const float*)d_in[2];
  const float* d_c = (const float*)d_in[3];
  const float* P[37];
  for (int i = 0; i < 37; ++i) P[i] = (const float*)d_in[4 + i];
  enum {
    bn1_g = 0, bn1_b, as1_W, as1_theta, as1_b, as2_W, as2_theta, as2_b,
    fc0_w, fc0_b, l40_w, l40_b, lstm_wih, lstm_whh, lstm_bih, lstm_bhh,
    fc11_w, fc11_b, fc12_w, fc12_b, fc13_w, fc13_b, fc2_w,
    b1_g, b1_b, b2_g, b2_b, b3_g, b3_b,
    bl1_w, bl1_b, bl2_w, bl2_b, bl3_w, bl3_b, bl4_w, bl4_b
  };
  const int* d_row = (const int*)d_in[41];
  const int* d_col = d_row + E_;

  // ---- workspace layout ----
  float* wsf = (float*)d_ws;
  const long long big = BTN_ * (long long)F_;   // 75,013,120 floats
  float* T1 = wsf;
  float* T2 = wsf + big;
  float* T3 = wsf + 2 * big;
  float* sp = wsf + 3 * big;
  float* ASC = sp; sp += N_;
  float* ASH = sp; sp += N_;
  float* DEG = sp; sp += N_;
  float* DNV = sp; sp += N_;
  float* NRM = sp; sp += E_;
  int* OFS = (int*)sp; sp += N_ + 1;
  int* CUR = (int*)sp; sp += N_;
  int* CSR = (int*)sp; sp += E_;
  float* CSW = sp; sp += E_;
  float* AWT = sp; sp += (long long)F_ * F_;
  sp = alignup(sp);
  unsigned short* PW = (unsigned short*)sp; sp += (6 * 12 * 512) / 2 + 16;   // Ntp=12 pad
  sp = alignup(sp);
  unsigned short* PH = (unsigned short*)sp; sp += (2 * 182 * 512) / 2 + 16;
  float* HST = sp; sp += (long long)N_ * 30;
  float* CST = sp; sp += (long long)N_ * 30;
  float* GTS = sp; sp += (long long)N_ * 120;
  float* M1 = sp; sp += (long long)B_ * N_ * 64;
  float* M2 = sp; sp += (long long)B_ * N_ * 32;
  float* HW = sp; sp += (long long)B_ * N_ * 50;
  float* SQN = sp; sp += N_;
  int* IDX = (int*)sp; sp += (long long)N_ * 6;
  float* WV = sp; sp += (long long)B_ * N_;
  float* BA = sp; sp += (long long)B_ * N_ * 100;
  float* BB = sp; sp += (long long)B_ * N_ * 60;
  float* BC = sp; sp += (long long)B_ * N_ * 20;
  float* BE = sp; sp += (long long)B_ * N_;
  float* ST2 = sp; sp += 2;

  // ---- output layout (tuple concatenated) ----
  float* out = (float*)d_out;
  float* oR = out;                                        // (B,T)      160
  float* oW = oR + (long long)B_ * T_;                    // (B,N,1)    11648
  float* oX = oW + (long long)B_ * N_;                    // (B,N,50)   582400
  float* oMU = oX + (long long)B_ * N_ * 50;              // (B,N,1)    11648
  float* oTMP = oMU + (long long)B_ * N_;                 // (B,N,N)    33918976
  float* oWADJ = oTMP + (long long)B_ * N_ * N_;          // (1,N,N)    8479744
  float* oPR = oWADJ + (long long)N_ * N_;                // (B,T,N)    465920

  // ---- bn1 folded into per-node scale/shift ----
  bn1_stats_kernel<<<N_, 256, 0, stream>>>(d_x, P[bn1_g], P[bn1_b], ASC, ASH);

  // ---- adjacency -> CSR ----
  fill_kernel<<<gsg(N_), 256, 0, stream>>>(DEG, 0.f, N_);
  deg_count_kernel<<<gsg(E_), 256, 0, stream>>>(d_col, DEG);
  dinv_kernel<<<gsg(N_), 256, 0, stream>>>(DEG, DNV);
  edge_norm_kernel<<<gsg(E_), 256, 0, stream>>>(d_row, d_col, DNV, NRM);
  scan_kernel<<<1, 1, 0, stream>>>(DEG, OFS);
  fill_kernel<<<gsg(N_), 256, 0, stream>>>((float*)CUR, 0.f, N_);
  csr_fill_kernel<<<gsg(E_), 256, 0, stream>>>(d_row, d_col, NRM, OFS, CUR, CSR, CSW);

  dim3 spmmGrid(N_, BT_);

  // ---- anti-conv layer 1 (bn fused on raw x) ----
  pack_b_kernel<<<gsg(6LL * 12 * 512), 256, 0, stream>>>(P[as1_theta], F_, F_, F_, 1, 6, 12, PW);
  launch_gemm(stream, d_x, F_, BTN_, F_, 6, PW, 12, F_, nullptr, T1, F_, 0, 0.f,
              ASC, ASH, N_, nullptr, nullptr);
  spmm_kernel<<<spmmGrid, 192, 0, stream>>>(T1, T2, OFS, CSR, CSW, DNV);
  make_aw_kernel<<<gsg((long long)F_ * F_), 256, 0, stream>>>(P[as1_W], AWT);
  pack_b_kernel<<<gsg(6LL * 12 * 512), 256, 0, stream>>>(AWT, F_, F_, F_, 1, 6, 12, PW);
  launch_gemm(stream, d_x, F_, BTN_, F_, 6, PW, 12, F_, P[as1_b], T1, F_, 2, 0.1f,
              ASC, ASH, N_, d_x, T2);

  // ---- anti-conv layer 2 (input now T1, plain) ----
  pack_b_kernel<<<gsg(6LL * 12 * 512), 256, 0, stream>>>(P[as2_theta], F_, F_, F_, 1, 6, 12, PW);
  launch_gemm(stream, T1, F_, BTN_, F_, 6, PW, 12, F_, nullptr, T2, F_, 0, 0.f,
              nullptr, nullptr, 0, nullptr, nullptr);
  spmm_kernel<<<spmmGrid, 192, 0, stream>>>(T2, T3, OFS, CSR, CSW, DNV);
  make_aw_kernel<<<gsg((long long)F_ * F_), 256, 0, stream>>>(P[as2_W], AWT);
  pack_b_kernel<<<gsg(6LL * 12 * 512), 256, 0, stream>>>(AWT, F_, F_, F_, 1, 6, 12, PW);
  launch_gemm(stream, T1, F_, BTN_, F_, 6, PW, 12, F_, P[as2_b], T2, F_, 2, 0.1f,
              nullptr, nullptr, 0, T1, T3);

  // ---- fc0: (F->50) + tanh ----
  pack_b_kernel<<<gsg(6LL * 4 * 512), 256, 0, stream>>>(P[fc0_w], F_, F_, 50, 1, 6, 4, PW);
  launch_gemm(stream, T2, F_, BTN_, F_, 6, PW, 4, 50, P[fc0_b], T1, 50, 1, 0.f,
              nullptr, nullptr, 0, nullptr, nullptr);

  // ---- l40 reduce over T -> xs (output x) ----
  l40_kernel<<<gsg((long long)B_ * N_ * 50), 256, 0, stream>>>(T1, P[l40_w], P[l40_b], oX);

  // ---- LSTM scan over B steps ----
  copy_kernel<<<gsg((long long)N_ * 30), 256, 0, stream>>>(HST, d_h, (long long)N_ * 30);
  copy_kernel<<<gsg((long long)N_ * 30), 256, 0, stream>>>(CST, d_c, (long long)N_ * 30);
  for (int b = 0; b < B_; ++b) {
    lstm_gates_kernel<<<gsg((long long)N_ * 120), 256, 0, stream>>>(
        oX + (long long)b * N_ * 50, HST, P[lstm_wih], P[lstm_whh], P[lstm_bih], P[lstm_bhh], GTS);
    lstm_update_kernel<<<gsg((long long)N_ * 30), 256, 0, stream>>>(GTS, HST, CST);
  }

  // ---- mu chain ----
  fc_kernel<<<gsg((long long)B_ * N_ * 64), 256, 0, stream>>>(oX, 50, P[fc11_w], P[fc11_b], M1, 64, (long long)B_ * N_, 1);
  fc_kernel<<<gsg((long long)B_ * N_ * 32), 256, 0, stream>>>(M1, 64, P[fc12_w], P[fc12_b], M2, 32, (long long)B_ * N_, 1);
  fc_kernel<<<gsg((long long)B_ * N_), 256, 0, stream>>>(M2, 32, P[fc13_w], P[fc13_b], oMU, 1, (long long)B_ * N_, 1);

  // ---- hw = xs @ fc2^T ----
  pack_b_kernel<<<gsg(2LL * 4 * 512), 256, 0, stream>>>(P[fc2_w], 50, 50, 50, 1, 2, 4, PW);
  launch_gemm(stream, oX, 50, (long long)B_ * N_, 50, 2, PW, 4, 50, nullptr, HW, 50, 0, 0.f,
              nullptr, nullptr, 0, nullptr, nullptr);

  // ---- temp[b] = tanh(hw_b @ hw_b^T) ----
  for (int b = 0; b < B_; ++b) {
    pack_b_kernel<<<gsg(2LL * 182 * 512), 256, 0, stream>>>(HW + (long long)b * N_ * 50, 50, 50, N_, 1, 2, 182, PH);
    launch_gemm(stream, HW + (long long)b * N_ * 50, 50, N_, 50, 2, PH, 182, N_, nullptr,
                oTMP + (long long)b * N_ * N_, N_, 1, 0.f, nullptr, nullptr, 0, nullptr, nullptr);
  }

  // ---- Wadj: 6-NN on H ----
  sqn_kernel<<<gsg(N_), 256, 0, stream>>>(HST, SQN);
  fill_kernel<<<gsg((long long)N_ * N_), 256, 0, stream>>>(oWADJ, 0.f, (long long)N_ * N_);
  topk_kernel<<<N_, 128, 0, stream>>>(HST, SQN, IDX, oWADJ);

  // ---- w / R ----
  wv_kernel<<<gsg((long long)B_ * N_), 256, 0, stream>>>(oMU, IDX, WV);
  wsoftmax_kernel<<<B_, 256, 0, stream>>>(WV, oW);
  r_reduce_kernel<<<B_ * T_, 256, 0, stream>>>(d_r, oW, oR);

  // ---- beta chain with global BN ----
  fc_kernel<<<gsg((long long)B_ * N_ * 100), 256, 0, stream>>>(oX, 50, P[bl1_w], P[bl1_b], BA, 100, (long long)B_ * N_, 0);
  fill_kernel<<<1, 256, 0, stream>>>(ST2, 0.f, 2);
  gstats_kernel<<<gsg((long long)B_ * N_ * 100), 256, 0, stream>>>(BA, (long long)B_ * N_ * 100, ST2);
  gbn_kernel<<<gsg((long long)B_ * N_ * 100), 256, 0, stream>>>(BA, (long long)B_ * N_ * 100, ST2, P[b1_g], P[b1_b], 1);
  fc_kernel<<<gsg((long long)B_ * N_ * 60), 256, 0, stream>>>(BA, 100, P[bl2_w], P[bl2_b], BB, 60, (long long)B_ * N_, 0);
  fill_kernel<<<1, 256, 0, stream>>>(ST2, 0.f, 2);
  gstats_kernel<<<gsg((long long)B_ * N_ * 60), 256, 0, stream>>>(BB, (long long)B_ * N_ * 60, ST2);
  gbn_kernel<<<gsg((long long)B_ * N_ * 60), 256, 0, stream>>>(BB, (long long)B_ * N_ * 60, ST2, P[b2_g], P[b2_b], 1);
  fc_kernel<<<gsg((long long)B_ * N_ * 20), 256, 0, stream>>>(BB, 60, P[bl3_w], P[bl3_b], BC, 20, (long long)B_ * N_, 0);
  fill_kernel<<<1, 256, 0, stream>>>(ST2, 0.f, 2);
  gstats_kernel<<<gsg((long long)B_ * N_ * 20), 256, 0, stream>>>(BC, (long long)B_ * N_ * 20, ST2);
  gbn_kernel<<<gsg((long long)B_ * N_ * 20), 256, 0, stream>>>(BC, (long long)B_ * N_ * 20, ST2, P[b3_g], P[b3_b], 1);
  fc_kernel<<<gsg((long long)B_ * N_), 256, 0, stream>>>(BC, 20, P[bl4_w], P[bl4_b], BE, 1, (long long)B_ * N_, 0);

  // ---- pre_r ----
  prer_kernel<<<gsg(BTN_), 256, 0, stream>>>(BE, oR, oPR);
}
